// PGRAAgg_79061757984921
// MI455X (gfx1250) — matrix operand order repaired and checked by reading, not compiled, not run on hardware
//
#include <hip/hip_runtime.h>
#include <hip/hip_bf16.h>
#include <math.h>

namespace {
constexpr int kB    = 32768;
constexpr int kK    = 50;
constexpr int kEmb  = 128;
constexpr int kRel  = 16;
constexpr int kWElems = 384 * 128;          // one weight matrix, elements
constexpr float kSlope = 0.2f;

typedef __attribute__((ext_vector_type(16))) __bf16 v16bf;
typedef __attribute__((ext_vector_type(8)))  __bf16 v8bf;
typedef __attribute__((ext_vector_type(8)))  float  v8f;

__device__ __forceinline__ unsigned short f2bf_bits(float f) {
  union { __bf16 b; unsigned short s; } u;
  u.b = static_cast<__bf16>(f);             // native v_cvt bf16 (RNE) on gfx1250
  return u.s;
}
__device__ __forceinline__ void cvt4(v16bf& d, int base, float4 f) {
  d[base + 0] = static_cast<__bf16>(f.x);
  d[base + 1] = static_cast<__bf16>(f.y);
  d[base + 2] = static_cast<__bf16>(f.z);
  d[base + 3] = static_cast<__bf16>(f.w);
}
__device__ __forceinline__ float sigmoid_fast(float x) {
  return 1.0f / (1.0f + __expf(-x));
}
__device__ __forceinline__ float tanh_fast(float x) {
  const float e = __expf(2.0f * x);
  return 1.0f - 2.0f / (e + 1.0f);
}
} // namespace

// One-shot: convert w_ih | w_hh (f32, row-major [384,128]) to bf16 in scratch.
__global__ __launch_bounds__(256) void prep_weights_kernel(
    const float* __restrict__ w_ih,
    const float* __restrict__ w_hh,
    unsigned short* __restrict__ wbf)   // [2*384*128] bf16: w_ih then w_hh
{
  const int i = (blockIdx.x * 256 + threadIdx.x) * 8;   // 8 elems per thread
  const float* src = (i < kWElems) ? (w_ih + i) : (w_hh + (i - kWElems));
  const float4 f0 = *(const float4*)(src);
  const float4 f1 = *(const float4*)(src + 4);
  uint4 p;
  p.x = ((unsigned int)f2bf_bits(f0.y) << 16) | f2bf_bits(f0.x);
  p.y = ((unsigned int)f2bf_bits(f0.w) << 16) | f2bf_bits(f0.z);
  p.z = ((unsigned int)f2bf_bits(f1.y) << 16) | f2bf_bits(f1.x);
  p.w = ((unsigned int)f2bf_bits(f1.w) << 16) | f2bf_bits(f1.z);
  *(uint4*)(wbf + i) = p;
}

// Fused: per-head masked attention over K neighbors + GRU mix with bf16 WMMA GEMMs.
__global__ __launch_bounds__(256) void pgra_fused_kernel(
    const float* __restrict__ self_vec,       // [B,128]
    const float* __restrict__ nb_vec,         // [B,50,128]
    const int*   __restrict__ trel,           // [B]
    const int*   __restrict__ nrel,           // [B,50]
    const float* __restrict__ rel_sim,        // [16,16]
    const unsigned char* __restrict__ mask,   // [B,50] (bool, 1 byte)
    const float* __restrict__ a_self,         // [8,16]
    const float* __restrict__ a_nb,           // [8,16]
    const unsigned short* __restrict__ wbf,   // bf16 weights: w_ih | w_hh
    const float* __restrict__ b_ih,           // [384]
    float*       __restrict__ out)            // [B,128]
{
  __shared__ unsigned short lds_nb[16 * kEmb];  // nb_vector tile, bf16

  const int tid  = threadIdx.x;
  const int lane = tid & 31;
  const int wave = tid >> 5;             // 0..7
  const int row0 = blockIdx.x * 16;      // 16 batch rows per block

  // ---------------- Phase 1: attention aggregation (1 wave -> 2 rows) ---------------
  {
    const int e = lane * 4;                                  // emb slice [e, e+4)
    const float4 as4 = *(const float4*)(a_self + e);
    const float4 an4 = *(const float4*)(a_nb + e);
    #pragma unroll
    for (int it = 0; it < 2; ++it) {
      const int r = wave * 2 + it;
      const int b = row0 + r;
      const float4 s4 = *(const float4*)(self_vec + (size_t)b * kEmb + e);
      // per-head (head = lane/4) dot(self, att_a_self) via 4-lane reduction
      float fs = s4.x * as4.x + s4.y * as4.y + s4.z * as4.z + s4.w * as4.w;
      fs += __shfl_xor(fs, 1);
      fs += __shfl_xor(fs, 2);

      const float* relrow = rel_sim + trel[b] * kRel;
      const int*   nr  = nrel + (size_t)b * kK;
      const unsigned char* mk = mask + (size_t)b * kK;
      const float* nbb = nb_vec + (size_t)b * kK * kEmb;

      float att[kK];
      #pragma unroll
      for (int k = 0; k < kK; ++k) {
        const float4 n4 = *(const float4*)(nbb + (size_t)k * kEmb + e);
        float d = n4.x * an4.x + n4.y * an4.y + n4.z * an4.z + n4.w * an4.w;
        d += __shfl_xor(d, 1);
        d += __shfl_xor(d, 2);
        float feat = d + fs + 1.0f;
        feat = (feat >= 0.0f) ? feat : kSlope * feat;       // leaky relu
        const float a = feat * relrow[nr[k]];
        att[k] = mk[k] ? a : -__builtin_inff();
      }
      // masked softmax over K (values replicated across the head's 4 lanes)
      float m = -__builtin_inff();
      #pragma unroll
      for (int k = 0; k < kK; ++k) m = fmaxf(m, att[k]);
      float s = 0.0f;
      #pragma unroll
      for (int k = 0; k < kK; ++k) {
        const float ev = (att[k] == -__builtin_inff()) ? 0.0f : __expf(att[k] - m);
        att[k] = ev; s += ev;
      }
      const float inv = 1.0f / s;
      #pragma unroll
      for (int k = 0; k < kK; ++k) att[k] *= inv;
      // weighted neighbor sum (re-read hits L2)
      float4 acc = {0.0f, 0.0f, 0.0f, 0.0f};
      #pragma unroll
      for (int k = 0; k < kK; ++k) {
        const float4 n4 = *(const float4*)(nbb + (size_t)k * kEmb + e);
        acc.x = fmaf(att[k], n4.x, acc.x);
        acc.y = fmaf(att[k], n4.y, acc.y);
        acc.z = fmaf(att[k], n4.z, acc.z);
        acc.w = fmaf(att[k], n4.w, acc.w);
      }
      const unsigned int p0 = ((unsigned int)f2bf_bits(acc.y) << 16) | f2bf_bits(acc.x);
      const unsigned int p1 = ((unsigned int)f2bf_bits(acc.w) << 16) | f2bf_bits(acc.z);
      *(uint2*)(&lds_nb[r * kEmb + e]) = make_uint2(p0, p1);
    }
  }
  __syncthreads();

  // ---------------- Phase 2: GRU GEMMs via v_wmma_f32_16x16x32_bf16 -----------------
  // Wave `wave` owns output channels [cbase, cbase+16). Needs N-tiles {c, 128+c, 256+c}
  // of both gi = nb @ w_ih^T and gh = self @ w_hh^T. K = 128 -> 4 WMMA steps.
  const int cbase = wave * 16;
  const int n     = lane & 15;     // A: row M; B: col N; D: col N
  const int hi16  = lane >> 4;     // 0: lanes 0-15, 1: lanes 16-31

  v8f zero = {0, 0, 0, 0, 0, 0, 0, 0};
  v8f acc_i[3] = {zero, zero, zero};   // ri, zi, hi
  v8f acc_h[3] = {zero, zero, zero};   // rh, zh, hh

  const __bf16* lnb    = (const __bf16*)lds_nb;
  const __bf16* wbf_ih = (const __bf16*)wbf;
  const __bf16* wbf_hh = (const __bf16*)(wbf + kWElems);

  #pragma unroll
  for (int ks = 0; ks < 4; ++ks) {
    // A fragment layout (16-bit A 16x32): lanes 0-15 hold K chunks {0..7,16..23},
    // lanes 16-31 hold {8..15,24..31}; element j -> K = chunk_base + (j<8 ? j : j+8).
    const int koffA = ks * 32 + hi16 * 8;
    const v8bf nlo = *(const v8bf*)(lnb + n * kEmb + koffA);
    const v8bf nhi = *(const v8bf*)(lnb + n * kEmb + koffA + 16);
    v16bf aN;
    #pragma unroll
    for (int j = 0; j < 8; ++j) { aN[j] = nlo[j]; aN[8 + j] = nhi[j]; }

    const float* sp = self_vec + (size_t)(row0 + n) * kEmb + koffA;
    v16bf aS;
    cvt4(aS, 0,  *(const float4*)(sp));
    cvt4(aS, 4,  *(const float4*)(sp + 4));
    cvt4(aS, 8,  *(const float4*)(sp + 16));
    cvt4(aS, 12, *(const float4*)(sp + 20));

    // B fragment layout (16-bit B 32x16): lane N = lane&15, element j -> K = 8*hi16 + (j<8 ? j : j+8).
    // Two aligned 16B vector loads per fragment.
    const int koffB = ks * 32 + hi16 * 8;
    #pragma unroll
    for (int t = 0; t < 3; ++t) {
      const int nrow = t * kEmb + cbase + n;   // weight row (gate-space column)
      {
        const v8bf blo = *(const v8bf*)(wbf_ih + (size_t)nrow * kEmb + koffB);
        const v8bf bhi = *(const v8bf*)(wbf_ih + (size_t)nrow * kEmb + koffB + 16);
        v16bf bF;
        #pragma unroll
        for (int j = 0; j < 8; ++j) { bF[j] = blo[j]; bF[8 + j] = bhi[j]; }
        acc_i[t] = __builtin_amdgcn_wmma_f32_16x16x32_bf16(
            false, aN, false, bF, (short)0, acc_i[t], false, false);
      }
      {
        const v8bf blo = *(const v8bf*)(wbf_hh + (size_t)nrow * kEmb + koffB);
        const v8bf bhi = *(const v8bf*)(wbf_hh + (size_t)nrow * kEmb + koffB + 16);
        v16bf bF;
        #pragma unroll
        for (int j = 0; j < 8; ++j) { bF[j] = blo[j]; bF[8 + j] = bhi[j]; }
        acc_h[t] = __builtin_amdgcn_wmma_f32_16x16x32_bf16(
            false, aS, false, bF, (short)0, acc_h[t], false, false);
      }
    }
  }

  // ------------- Epilogue: gates + blend. C/D layout: M = v + 8*hi16, N = lane&15 ----
  const int col = cbase + n;
  const float bri = b_ih[col];
  const float bzi = b_ih[kEmb + col];
  const float bhi = b_ih[2 * kEmb + col];
  #pragma unroll
  for (int v = 0; v < 8; ++v) {
    const int row = row0 + v + hi16 * 8;
    const float ri = acc_i[0][v] + bri;
    const float zi = acc_i[1][v] + bzi;
    const float hi = acc_i[2][v] + bhi;
    const float rh = acc_h[0][v];
    const float zh = acc_h[1][v];
    const float hh = acc_h[2][v];
    const float rg = sigmoid_fast(ri + rh);
    const float zg = sigmoid_fast(zi + zh);
    const float hc = tanh_fast(hi + hh * rg);
    const float sv = self_vec[(size_t)row * kEmb + col];
    out[(size_t)row * kEmb + col] = (1.0f - zg) * sv + zg * hc;
  }
}

extern "C" void kernel_launch(void* const* d_in, const int* in_sizes, int n_in,
                              void* d_out, int out_size, void* d_ws, size_t ws_size,
                              hipStream_t stream) {
  (void)in_sizes; (void)n_in; (void)out_size; (void)ws_size;
  const float* self_vec = (const float*)d_in[0];
  const float* nb_vec   = (const float*)d_in[1];
  const int*   trel     = (const int*)d_in[2];
  const int*   nrel     = (const int*)d_in[3];
  const float* rel_sim  = (const float*)d_in[4];
  const unsigned char* mask = (const unsigned char*)d_in[5];
  const float* a_self   = (const float*)d_in[6];
  const float* a_nb     = (const float*)d_in[7];
  const float* w_ih     = (const float*)d_in[8];
  const float* b_ih     = (const float*)d_in[9];
  const float* w_hh     = (const float*)d_in[10];
  unsigned short* wbf   = (unsigned short*)d_ws;   // 192 KB bf16 weight stash

  // 2*384*128 elements, 8 per thread, 256 threads/block -> 48 blocks.
  prep_weights_kernel<<<(2 * kWElems) / (256 * 8), 256, 0, stream>>>(w_ih, w_hh, wbf);

  pgra_fused_kernel<<<kB / 16, 256, 0, stream>>>(
      self_vec, nb_vec, trel, nrel, rel_sim, mask, a_self, a_nb,
      wbf, b_ih, (float*)d_out);
}
